// GeniePathLayer_15917148799862
// MI455X (gfx1250) — matrix operand order repaired and checked by reading, not compiled, not run on hardware
//
#include <hip/hip_runtime.h>

typedef __attribute__((ext_vector_type(16))) __bf16 v16bf;
typedef __attribute__((ext_vector_type(8)))  float  v8f;

#define DMODEL 64
#define GATES  256
#define HIDDEN 64

__device__ __forceinline__ float sigmoidf_(float x) { return 1.0f / (1.0f + __expf(-x)); }

__device__ __forceinline__ void atomic_max_float(float* addr, float val) {
  if (val >= 0.0f) atomicMax((int*)addr, __float_as_int(val));
  else             atomicMin((unsigned int*)addr, __float_as_uint(val));
}

__device__ __forceinline__ void edge_nodes(const long long* __restrict__ ei,
                                           long long E, long long e, int* s, int* d) {
  if (e < E) { *s = (int)ei[e]; *d = (int)ei[E + e]; }
  else       { int n = (int)(e - E); *s = n; *d = n; }   // implicit self loops
}

// pack two float4 (8 contiguous f32) into bf16 fragment elems [off..off+7], scaled
__device__ __forceinline__ void put8(v16bf& a, int off, float4 u, float4 v, float sc) {
  a[off + 0] = (__bf16)(u.x * sc); a[off + 1] = (__bf16)(u.y * sc);
  a[off + 2] = (__bf16)(u.z * sc); a[off + 3] = (__bf16)(u.w * sc);
  a[off + 4] = (__bf16)(v.x * sc); a[off + 5] = (__bf16)(v.y * sc);
  a[off + 6] = (__bf16)(v.z * sc); a[off + 7] = (__bf16)(v.w * sc);
}

// ---- init: agg = 0, m = -inf, s = 0 -------------------------------------------------
__global__ void init_kernel(float* __restrict__ agg, float* __restrict__ m,
                            float* __restrict__ ssum, int N) {
  long long i = (long long)blockIdx.x * blockDim.x + threadIdx.x;
  if (i < (long long)N * DMODEL) agg[i] = 0.0f;
  if (i < N) { m[i] = __int_as_float(0xff800000); ssum[i] = 0.0f; }
}

// ---- GAT projection: xw = x @ W_gat  (N x 64) @ (64 x 64), bf16 WMMA ---------------
__global__ __launch_bounds__(256) void gat_gemm_kernel(
    const float* __restrict__ x, const float* __restrict__ W,
    float* __restrict__ xw, int N) {
  int tid  = threadIdx.x;
  int lane = tid & 31;
  int wave = tid >> 5;                 // 8 waves: 2 row tiles x 4 col tiles
  int rowBase = blockIdx.x * 32 + (wave >> 2) * 16;
  int colBase = (wave & 3) * 16;

  int rIn = lane & 15;
  int kg  = lane >> 4;
  int row = rowBase + rIn;
  float asc = (row < N) ? 1.0f : 0.0f;          // zero padding rows branchlessly
  int rowC  = (row < N) ? row : 0;              // clamped, always-valid address
  const float4* xp = (const float4*)(x + (long long)rowC * DMODEL);

  int col   = colBase + (lane & 15);
  int khalf = (lane >> 4) * 16;

  v8f acc = {};
#pragma unroll
  for (int s = 0; s < 2; ++s) {                 // K = 64 -> 2x K32 steps
    v16bf a, b;
    int q0 = (32 * s + kg * 8) >> 2;            // j 0..7  : K = 32s + kg*8 + j
    int q1 = (32 * s + 16 + kg * 8) >> 2;       // j 8..15 : K = 32s + 16 + kg*8 + (j-8)
    put8(a, 0, xp[q0], xp[q0 + 1], asc);
    put8(a, 8, xp[q1], xp[q1 + 1], asc);
#pragma unroll
    for (int e = 0; e < 16; ++e) {              // lane-coalesced, k-strided
      int k = 32 * s + kg * 8 + (e & 7) + (e >> 3) * 16;
      b[e] = (__bf16)W[k * DMODEL + col];
    }
    acc = __builtin_amdgcn_wmma_f32_16x16x32_bf16(false, a, false, b,
                                                  (short)0, acc, false, false);
  }

  int mrow = rowBase + (lane >> 4) * 8;
  float* op = xw + (long long)mrow * DMODEL + col;  // single base; imm offsets below
  if (mrow + 7 < N) {                               // fast path: whole 8-row strip in range
#pragma unroll
    for (int r = 0; r < 8; ++r) op[r * DMODEL] = acc[r];
  } else {                                          // boundary strip only
#pragma unroll
    for (int r = 0; r < 8; ++r) if (mrow + r < N) op[r * DMODEL] = acc[r];
  }
}

// ---- attention coefficients per node ------------------------------------------------
__global__ void att_kernel(const float* __restrict__ xw,
                           const float* __restrict__ att_s, const float* __restrict__ att_d,
                           float* __restrict__ a_src, float* __restrict__ a_dst, int N) {
  int n = blockIdx.x * blockDim.x + threadIdx.x;
  if (n >= N) return;
  const float4* r  = (const float4*)(xw + (long long)n * DMODEL);
  const float4* as = (const float4*)att_s;
  const float4* ad = (const float4*)att_d;
  float ss = 0.0f, dd = 0.0f;
#pragma unroll
  for (int q = 0; q < DMODEL / 4; ++q) {
    float4 v = r[q], s4 = as[q], d4 = ad[q];
    ss += v.x * s4.x + v.y * s4.y + v.z * s4.z + v.w * s4.w;
    dd += v.x * d4.x + v.y * d4.y + v.z * d4.z + v.w * d4.w;
  }
  a_src[n] = ss; a_dst[n] = dd;
}

// ---- edge pass 1: segment max of leaky-relu logits ---------------------------------
__global__ void edge_max_kernel(const long long* __restrict__ ei, long long E, long long Etot,
                                const float* __restrict__ a_src, const float* __restrict__ a_dst,
                                float* __restrict__ m) {
  long long e = (long long)blockIdx.x * blockDim.x + threadIdx.x;
  if (e >= Etot) return;
  int s, d; edge_nodes(ei, E, e, &s, &d);
  float v = a_src[s] + a_dst[d];
  v = v > 0.0f ? v : 0.2f * v;
  atomic_max_float(&m[d], v);
}

// ---- edge pass 2: segment sum of exp(logit - max) ----------------------------------
__global__ void edge_sum_kernel(const long long* __restrict__ ei, long long E, long long Etot,
                                const float* __restrict__ a_src, const float* __restrict__ a_dst,
                                const float* __restrict__ m, float* __restrict__ ssum) {
  long long e = (long long)blockIdx.x * blockDim.x + threadIdx.x;
  if (e >= Etot) return;
  int s, d; edge_nodes(ei, E, e, &s, &d);
  float v = a_src[s] + a_dst[d];
  v = v > 0.0f ? v : 0.2f * v;
  atomicAdd(&ssum[d], __expf(v - m[d]));
}

// ---- edge pass 3: scatter alpha-weighted messages (64 lanes per edge) --------------
__global__ void edge_scatter_kernel(const long long* __restrict__ ei, long long E, long long Etot,
                                    const float* __restrict__ a_src, const float* __restrict__ a_dst,
                                    const float* __restrict__ m, const float* __restrict__ ssum,
                                    const float* __restrict__ xw, float* __restrict__ agg) {
  long long t = (long long)blockIdx.x * blockDim.x + threadIdx.x;
  long long e = t >> 6;
  int j = (int)(t & 63);
  if (e >= Etot) return;
  int s, d; edge_nodes(ei, E, e, &s, &d);
  float v = a_src[s] + a_dst[d];
  v = v > 0.0f ? v : 0.2f * v;
  float alpha = __expf(v - m[d]) / (ssum[d] + 1e-16f);
  atomicAdd(&agg[(long long)d * DMODEL + j], xw[(long long)s * DMODEL + j] * alpha);
}

// ---- xt = tanh(agg + bias) ----------------------------------------------------------
__global__ void xt_kernel(const float* __restrict__ agg, const float* __restrict__ bias,
                          float* __restrict__ xt, int N) {
  long long i = (long long)blockIdx.x * blockDim.x + threadIdx.x;
  if (i >= (long long)N * DMODEL) return;
  xt[i] = tanhf(agg[i] + bias[(int)(i & 63)]);
}

// ---- fused LSTM: gates = xt@W_ih^T + h@W_hh^T via WMMA, then activations ------------
__global__ __launch_bounds__(512) void lstm_kernel(
    const float* __restrict__ xt, const float* __restrict__ h,
    const float* __restrict__ cprev, const float* __restrict__ Wih,
    const float* __restrict__ Whh, float* __restrict__ out, int N) {
  __shared__ float g_s[16 * GATES];   // 16 rows x 256 gate cols, 16 KB
  int tid  = threadIdx.x;
  int lane = tid & 31;
  int wave = tid >> 5;                // 16 waves -> 16 col tiles covering 256 gates
  int rowBase = blockIdx.x * 16;
  int colBase = wave * 16;

  int rIn = lane & 15;
  int kg  = lane >> 4;
  int row = rowBase + rIn;
  float asc = (row < N) ? 1.0f : 0.0f;
  int rowC  = (row < N) ? row : 0;
  const float4* xp = (const float4*)(xt + (long long)rowC * DMODEL);
  const float4* hp = (const float4*)(h  + (long long)rowC * HIDDEN);

  int col   = colBase + (lane & 15);
  int khalf = (lane >> 4) * 16;
  const float4* bip = (const float4*)(Wih + col * DMODEL);  // W_ih[g][k], contiguous in k
  const float4* bhp = (const float4*)(Whh + col * HIDDEN);

  v8f acc = {};
#pragma unroll
  for (int s2 = 0; s2 < 2; ++s2) {
    v16bf a, b;
    int q0 = (32 * s2 + kg * 8) >> 2;
    int q1 = (32 * s2 + 16 + kg * 8) >> 2;
    int qb = (32 * s2 + khalf) >> 2;            // B: 16 contiguous k per lane

    // xt @ W_ih^T
    put8(a, 0, xp[qb], xp[qb + 1], asc);
    put8(a, 8, xp[qb + 2], xp[qb + 3], asc);
    put8(b, 0, bip[qb], bip[qb + 1], 1.0f);
    put8(b, 8, bip[qb + 2], bip[qb + 3], 1.0f);
    acc = __builtin_amdgcn_wmma_f32_16x16x32_bf16(false, a, false, b,
                                                  (short)0, acc, false, false);
    // h @ W_hh^T
    put8(a, 0, hp[qb], hp[qb + 1], asc);
    put8(a, 8, hp[qb + 2], hp[qb + 3], asc);
    put8(b, 0, bhp[qb], bhp[qb + 1], 1.0f);
    put8(b, 8, bhp[qb + 2], bhp[qb + 3], 1.0f);
    acc = __builtin_amdgcn_wmma_f32_16x16x32_bf16(false, a, false, b,
                                                  (short)0, acc, false, false);
  }
  int mrow = (lane >> 4) * 8;
#pragma unroll
  for (int r = 0; r < 8; ++r) g_s[(mrow + r) * GATES + col] = acc[r];
  __syncthreads();

  long long NH = (long long)N * HIDDEN;
  bool fullBlock = (rowBase + 16) <= N;          // block-uniform -> scalar branch
#pragma unroll
  for (int it = 0; it < 2; ++it) {               // 1024 (row, hidden) outputs / 512 threads
    int idx  = tid + it * 512;
    int rrow = idx >> 6;
    int j    = idx & 63;
    int n    = rowBase + rrow;
    if (fullBlock || n < N) {
      float ig = g_s[rrow * GATES + j];
      float fg = g_s[rrow * GATES + 64 + j];
      float gg = g_s[rrow * GATES + 128 + j];
      float og = g_s[rrow * GATES + 192 + j];
      float cp = cprev[(long long)n * HIDDEN + j];
      float cn = sigmoidf_(fg) * cp + sigmoidf_(ig) * tanhf(gg);
      float hn = sigmoidf_(og) * tanhf(cn);
      long long base = (long long)n * HIDDEN + j;
      out[base]          = hn;   // h_new
      out[NH + base]     = hn;   // h_new[None]
      out[2 * NH + base] = cn;   // c_new[None]
    }
  }
}

extern "C" void kernel_launch(void* const* d_in, const int* in_sizes, int n_in,
                              void* d_out, int out_size, void* d_ws, size_t ws_size,
                              hipStream_t stream) {
  const float*     x     = (const float*)d_in[0];
  const long long* ei    = (const long long*)d_in[1];   // int64 edge_index [2,E]
  const float*     h     = (const float*)d_in[2];
  const float*     c     = (const float*)d_in[3];
  const float*     Wgat  = (const float*)d_in[4];
  const float*     att_s = (const float*)d_in[5];
  const float*     att_d = (const float*)d_in[6];
  const float*     bias  = (const float*)d_in[7];
  const float*     Wih   = (const float*)d_in[8];
  const float*     Whh   = (const float*)d_in[9];

  int N        = in_sizes[0] / DMODEL;
  long long E  = in_sizes[1] / 2;
  long long Et = E + N;

  char* wsp = (char*)d_ws;
  auto carve = [&](size_t nbytes) -> float* {
    float* p = (float*)wsp;
    wsp += (nbytes + 255) & ~(size_t)255;
    return p;
  };
  float* xw    = carve((size_t)N * DMODEL * sizeof(float));
  float* agg   = carve((size_t)N * DMODEL * sizeof(float));
  float* xt    = carve((size_t)N * DMODEL * sizeof(float));
  float* a_src = carve((size_t)N * sizeof(float));
  float* a_dst = carve((size_t)N * sizeof(float));
  float* m     = carve((size_t)N * sizeof(float));
  float* ssum  = carve((size_t)N * sizeof(float));

  long long NW = (long long)N * DMODEL;
  init_kernel        <<<(int)((NW + 255) / 256), 256, 0, stream>>>(agg, m, ssum, N);
  gat_gemm_kernel    <<<(N + 31) / 32, 256, 0, stream>>>(x, Wgat, xw, N);
  att_kernel         <<<(N + 255) / 256, 256, 0, stream>>>(xw, att_s, att_d, a_src, a_dst, N);
  edge_max_kernel    <<<(int)((Et + 255) / 256), 256, 0, stream>>>(ei, E, Et, a_src, a_dst, m);
  edge_sum_kernel    <<<(int)((Et + 255) / 256), 256, 0, stream>>>(ei, E, Et, a_src, a_dst, m, ssum);
  edge_scatter_kernel<<<(int)((Et * 64 + 255) / 256), 256, 0, stream>>>(ei, E, Et, a_src, a_dst,
                                                                        m, ssum, xw, agg);
  xt_kernel          <<<(int)((NW + 255) / 256), 256, 0, stream>>>(agg, bias, xt, N);
  lstm_kernel        <<<(N + 15) / 16, 512, 0, stream>>>(xt, h, c, Wih, Whh, (float*)d_out, N);
}
